// ChebNet_90890097918699
// MI455X (gfx1250) — matrix operand order repaired and checked by reading, not compiled, not run on hardware
//
#include <hip/hip_runtime.h>
#include <hip/hip_bf16.h>

typedef __attribute__((ext_vector_type(16))) __bf16 v16bf;
typedef __attribute__((ext_vector_type(8)))  __bf16 v8bf;
typedef __attribute__((ext_vector_type(8)))  float  v8f;

constexpr int B  = 48;
constexpr int N  = 512;
constexpr int C  = 64;
constexpr int CO = 64;
constexpr int K1 = 5;
constexpr float EPSF = 1e-9f;
constexpr float LOG2E = 1.44269504088896340736f;

// ---------- small helpers ----------

union V16U { v16bf v; unsigned u[8]; };
union V8U  { v8bf  v; uint4   u;    };

// native f32 -> bf16 conversion (expect v_cvt_pk_bf16_f32-class lowering)
__device__ inline v16bf pack16(const float* f) {
  v16bf t;
#pragma unroll
  for (int e = 0; e < 16; ++e) t[e] = (__bf16)f[e];
  return t;
}

// negate a packed bf16x16 operand with 8 XORs on the dwords
__device__ inline v16bf negbf(v16bf x) {
  V16U t;
  t.v = x;
#pragma unroll
  for (int e = 0; e < 8; ++e) t.u[e] ^= 0x80008000u;
  return t.v;
}

__device__ inline uint4 packD8(v8f a) {
  V8U t;
#pragma unroll
  for (int e = 0; e < 8; ++e) t.v[e] = (__bf16)a[e];
  return t.u;
}

__device__ inline v8f vzero8() {
  v8f z;
#pragma unroll
  for (int e = 0; e < 8; ++e) z[e] = 0.0f;
  return z;
}

// Load the 16 f32 elements this lane owns for a 16x32 bf16 A-operand tile.
// lane half h (0/1): elements 0..15 <- [16h, 16h+16)
__device__ inline void loadA16(const float* base, int h, float* d) {
  const float4 a0 = *(const float4*)(base + 16 * h);
  const float4 a1 = *(const float4*)(base + 16 * h + 4);
  const float4 a2 = *(const float4*)(base + 16 * h + 8);
  const float4 a3 = *(const float4*)(base + 16 * h + 12);
  d[0] = a0.x;  d[1] = a0.y;  d[2] = a0.z;  d[3] = a0.w;
  d[4] = a1.x;  d[5] = a1.y;  d[6] = a1.z;  d[7] = a1.w;
  d[8] = a2.x;  d[9] = a2.y;  d[10] = a2.z; d[11] = a2.w;
  d[12] = a3.x; d[13] = a3.y; d[14] = a3.z; d[15] = a3.w;
}

__device__ inline v8f wmma_bf(v16bf a, v16bf b, v8f c) {
  return __builtin_amdgcn_wmma_f32_16x16x32_bf16(false, a, false, b, (short)0, c,
                                                 false, false);
}

// shared modReLU math so stats + main kernels agree bit-for-bit
__device__ inline void modrelu_pair(float scr, float sci, float mb,
                                    float& vr, float& vi) {
  float mag = __builtin_amdgcn_sqrtf(scr * scr + sci * sci);
  float sc  = fmaxf(mag + mb, 0.0f) * __builtin_amdgcn_rcpf(mag + EPSF);
  vr = scr * sc;
  vi = sci * sc;
}

// ---------- kernel 1: W (K,C,O) f32 -> WT (K,O,C) bf16, for B-operand loads ----------

__global__ __launch_bounds__(256) void kern_wt(const float* __restrict__ Wr,
                                               const float* __restrict__ Wi,
                                               unsigned short* __restrict__ WrT,
                                               unsigned short* __restrict__ WiT) {
  int t = blockIdx.x * 256 + threadIdx.x;   // t = ((k*CO + o)*C + c)
  if (t >= K1 * CO * C) return;
  int c = t % C;
  int o = (t / C) % CO;
  int k = t / (C * CO);
  __bf16 br = (__bf16)Wr[(k * C + c) * CO + o];
  __bf16 bi = (__bf16)Wi[(k * C + c) * CO + o];
  WrT[t] = __builtin_bit_cast(unsigned short, br);
  WiT[t] = __builtin_bit_cast(unsigned short, bi);
}

// ---------- kernel 2: attention score vectors si/sj per (b,n) ----------

__global__ __launch_bounds__(256) void kern_s(const float* __restrict__ Xr,
                                              const float* __restrict__ Xi,
                                              const float* __restrict__ War,
                                              const float* __restrict__ Wai,
                                              float* __restrict__ sir, float* __restrict__ sii,
                                              float* __restrict__ sjr, float* __restrict__ sji) {
  int t = blockIdx.x * 256 + threadIdx.x;   // (b*N + n)
  if (t >= B * N) return;
  const float* xr = Xr + (size_t)t * C;
  const float* xi = Xi + (size_t)t * C;
  float vir = 0.f, vii = 0.f, vjr = 0.f, vji = 0.f;
#pragma unroll 8
  for (int c = 0; c < C; ++c) {
    float xrv = xr[c], xiv = xi[c];
    float w1r = War[c], w2r = War[C + c];
    float w1i = Wai[c], w2i = Wai[C + c];
    vir = fmaf(xrv, w1r, vir); vir = fmaf(-xiv, w1i, vir);
    vii = fmaf(xrv, w1i, vii); vii = fmaf(xiv, w1r, vii);
    vjr = fmaf(xrv, w2r, vjr); vjr = fmaf(-xiv, w2i, vjr);
    vji = fmaf(xrv, w2i, vji); vji = fmaf(xiv, w2r, vji);
  }
  sir[t] = vir; sii[t] = vii; sjr[t] = vjr; sji[t] = vji;
}

// ---------- kernel 3: per-row softmax stats (max, 1/sum) for both r and i ----------

__global__ __launch_bounds__(256) void kern_stats(const float* __restrict__ sir,
                                                  const float* __restrict__ sii,
                                                  const float* __restrict__ sjr,
                                                  const float* __restrict__ sji,
                                                  const float* __restrict__ bar_p,
                                                  const float* __restrict__ bai_p,
                                                  const float* __restrict__ mb_p,
                                                  float* __restrict__ maxr, float* __restrict__ ivsr,
                                                  float* __restrict__ maxi, float* __restrict__ ivsi) {
  int wave = (blockIdx.x * 256 + threadIdx.x) >> 5;   // one wave per (b,i) row
  int lane = threadIdx.x & 31;
  if (wave >= B * N) return;
  int b = wave / N;
  float bar = *bar_p, bai = *bai_p, mb = *mb_p;
  float s_ir = sir[wave] + bar, s_ii = sii[wave] + bai;
  const float* pjr = sjr + (size_t)b * N;
  const float* pji = sji + (size_t)b * N;
  float vr[16], vi[16];
  float mr = -3.4e38f, mi = -3.4e38f;
#pragma unroll
  for (int t = 0; t < 16; ++t) {
    int j = t * 32 + lane;
    modrelu_pair(s_ir + pjr[j], s_ii + pji[j], mb, vr[t], vi[t]);
    mr = fmaxf(mr, vr[t]);
    mi = fmaxf(mi, vi[t]);
  }
#pragma unroll
  for (int m = 16; m >= 1; m >>= 1) {
    mr = fmaxf(mr, __shfl_xor(mr, m, 32));
    mi = fmaxf(mi, __shfl_xor(mi, m, 32));
  }
  float sr = 0.f, si = 0.f;
#pragma unroll
  for (int t = 0; t < 16; ++t) {
    sr += __builtin_amdgcn_exp2f((vr[t] - mr) * LOG2E);
    si += __builtin_amdgcn_exp2f((vi[t] - mi) * LOG2E);
  }
#pragma unroll
  for (int m = 16; m >= 1; m >>= 1) {
    sr += __shfl_xor(sr, m, 32);
    si += __shfl_xor(si, m, 32);
  }
  if (lane == 0) {
    maxr[wave] = mr; ivsr[wave] = 1.0f / sr;
    maxi[wave] = mi; ivsi[wave] = 1.0f / si;
  }
}

// ---------- kernel 4: P1/P2 = complex combos of X@W[k], bf16, stored o-major ----------
// P1[k][b](j,o) = Xr@Wr - Xi@Wi ; P2 = Xi@Wr + Xr@Wi ; stored as [k][b][o][j] bf16

__global__ __launch_bounds__(256) void kern_p(const float* __restrict__ Xr,
                                              const float* __restrict__ Xi,
                                              const unsigned short* __restrict__ WrT,
                                              const unsigned short* __restrict__ WiT,
                                              unsigned short* __restrict__ P1T,
                                              unsigned short* __restrict__ P2T) {
  int wave = (blockIdx.x * 256 + threadIdx.x) >> 5;   // one wave per (k,b,jtile16)
  int lane = threadIdx.x & 31;
  const int JT = N / 16;
  if (wave >= K1 * B * JT) return;
  int jt = wave % JT;
  int b  = (wave / JT) % B;
  int k  = wave / (JT * B);
  int j0 = jt * 16;
  int r = lane & 15, h = lane >> 4;

  v8f acc1[4], acc2[4];
#pragma unroll
  for (int ot = 0; ot < 4; ++ot) { acc1[ot] = vzero8(); acc2[ot] = vzero8(); }

#pragma unroll
  for (int c2 = 0; c2 < 2; ++c2) {
    float fr[16], fi[16];
    const float* xr = Xr + ((size_t)(b * N + j0 + r)) * C + c2 * 32;
    const float* xi = Xi + ((size_t)(b * N + j0 + r)) * C + c2 * 32;
    loadA16(xr, h, fr);
    loadA16(xi, h, fi);
    v16bf Ar  = pack16(fr);
    v16bf Ai  = pack16(fi);
    v16bf Ain = negbf(Ai);
#pragma unroll
    for (int ot = 0; ot < 4; ++ot) {
      size_t wb = (size_t)(k * CO + ot * 16 + r) * C + c2 * 32 + 16 * h;
      v16bf Bwr = *(const v16bf*)(WrT + wb);
      v16bf Bwi = *(const v16bf*)(WiT + wb);
      acc1[ot] = wmma_bf(Ar,  Bwr, acc1[ot]);   // + Xr Wr
      acc1[ot] = wmma_bf(Ain, Bwi, acc1[ot]);   // - Xi Wi
      acc2[ot] = wmma_bf(Ai,  Bwr, acc2[ot]);   // + Xi Wr
      acc2[ot] = wmma_bf(Ar,  Bwi, acc2[ot]);   // + Xr Wi
    }
  }
  // transposed bf16 store: lane owns column o = ot*16+r, rows j0+8h+v (contiguous)
#pragma unroll
  for (int ot = 0; ot < 4; ++ot) {
    size_t dst = (size_t)((k * B + b) * CO + ot * 16 + r) * N + j0 + 8 * h;
    *(uint4*)(P1T + dst) = packD8(acc1[ot]);
    *(uint4*)(P2T + dst) = packD8(acc2[ot]);
  }
}

// ---------- kernel 5: fused attention * lap * P -> out (the 32-GFLOP matmul) ----------

__global__ __launch_bounds__(128) void kern_main(
    const float* __restrict__ lapr_g, const float* __restrict__ lapi_g,
    const float* __restrict__ sir, const float* __restrict__ sii,
    const float* __restrict__ sjr, const float* __restrict__ sji,
    const float* __restrict__ maxr, const float* __restrict__ ivsr,
    const float* __restrict__ maxi, const float* __restrict__ ivsi,
    const float* __restrict__ bar_p, const float* __restrict__ bai_p,
    const float* __restrict__ mb_p,
    const unsigned short* __restrict__ P1T, const unsigned short* __restrict__ P2T,
    float* __restrict__ outr, float* __restrict__ outi) {
  int b    = blockIdx.x;
  int wave = threadIdx.x >> 5;
  int lane = threadIdx.x & 31;
  int i0   = (blockIdx.y * 4 + wave) * 16;    // 16-row i-tile per wave
  int r = lane & 15, h = lane >> 4;

  int row = b * N + i0 + r;
  float bar = *bar_p, bai = *bai_p, mb = *mb_p;
  float s_ir = sir[row] + bar, s_ii = sii[row] + bai;   // fold bias once
  // fold softmax max & 1/sum into one exp2 argument offset:
  // a = exp2(v*log2e + c) with c = log2(invsum) - max*log2e
  float c_r = __builtin_amdgcn_logf(ivsr[row]) * 1.0f - maxr[row] * LOG2E;
  float c_i = __builtin_amdgcn_logf(ivsi[row]) * 1.0f - maxi[row] * LOG2E;

  v8f accR[4], accI[4];
#pragma unroll
  for (int ct = 0; ct < 4; ++ct) { accR[ct] = vzero8(); accI[ct] = vzero8(); }

  const float* sjr_b = sjr + (size_t)b * N;
  const float* sji_b = sji + (size_t)b * N;

  for (int jt = 0; jt < 16; ++jt) {
    int j0 = jt * 32;
    if (jt < 15) {  // prefetch next j-tile of lap (k=0) -> global_prefetch_b8
      __builtin_prefetch(lapr_g + ((size_t)(b * K1) * N + i0 + r) * N + j0 + 32, 0, 1);
      __builtin_prefetch(lapi_g + ((size_t)(b * K1) * N + i0 + r) * N + j0 + 32, 0, 1);
    }
    // attention weights for this lane's 16 (row r, j) positions; reused over K=5
    float sjrv[16], sjiv[16], arr[16], aiv[16];
    loadA16(sjr_b + j0, h, sjrv);
    loadA16(sji_b + j0, h, sjiv);
#pragma unroll
    for (int e = 0; e < 16; ++e) {
      float vr, vi;
      modrelu_pair(s_ir + sjrv[e], s_ii + sjiv[e], mb, vr, vi);
      arr[e] = __builtin_amdgcn_exp2f(fmaf(vr, LOG2E, c_r));
      aiv[e] = __builtin_amdgcn_exp2f(fmaf(vi, LOG2E, c_i));
    }
#pragma unroll
    for (int k = 0; k < K1; ++k) {
      size_t lbase = ((size_t)(b * K1 + k) * N + i0 + r) * N + j0;
      float flr[16], fli[16];
      loadA16(lapr_g + lbase, h, flr);
      loadA16(lapi_g + lbase, h, fli);
      float Lr[16], Li[16];
#pragma unroll
      for (int e = 0; e < 16; ++e) {
        Lr[e] = flr[e] * arr[e] - fli[e] * aiv[e];
        Li[e] = flr[e] * aiv[e] + fli[e] * arr[e];
      }
      v16bf Ar  = pack16(Lr);
      v16bf Ai  = pack16(Li);
      v16bf Ain = negbf(Ai);
      const unsigned short* p1 = P1T + (size_t)((k * B + b) * CO) * N;
      const unsigned short* p2 = P2T + (size_t)((k * B + b) * CO) * N;
#pragma unroll
      for (int ct = 0; ct < 4; ++ct) {
        size_t off = (size_t)(ct * 16 + r) * N + j0 + 16 * h;
        v16bf B1 = *(const v16bf*)(p1 + off);
        v16bf B2 = *(const v16bf*)(p2 + off);
        accR[ct] = wmma_bf(Ar,  B1, accR[ct]);   // + L_r P1
        accR[ct] = wmma_bf(Ain, B2, accR[ct]);   // - L_i P2
        accI[ct] = wmma_bf(Ar,  B2, accI[ct]);   // + L_r P2
        accI[ct] = wmma_bf(Ai,  B1, accI[ct]);   // + L_i P1
      }
    }
  }

  // D layout: lane -> column o = ct*16 + r, rows i0 + 8h + v
  size_t obase = ((size_t)b * N + i0 + 8 * h) * CO;
#pragma unroll
  for (int ct = 0; ct < 4; ++ct) {
    int o = ct * 16 + r;
#pragma unroll
    for (int v = 0; v < 8; ++v) {
      outr[obase + (size_t)v * CO + o] = accR[ct][v];
      outi[obase + (size_t)v * CO + o] = accI[ct][v];
    }
  }
}

// ---------- launch ----------

extern "C" void kernel_launch(void* const* d_in, const int* in_sizes, int n_in,
                              void* d_out, int out_size, void* d_ws, size_t ws_size,
                              hipStream_t stream) {
  const float* Xr   = (const float*)d_in[0];
  const float* Xi   = (const float*)d_in[1];
  const float* lapr = (const float*)d_in[2];
  const float* lapi = (const float*)d_in[3];
  const float* War  = (const float*)d_in[4];
  const float* Wai  = (const float*)d_in[5];
  const float* bar  = (const float*)d_in[6];
  const float* bai  = (const float*)d_in[7];
  const float* mb   = (const float*)d_in[8];
  const float* Wre  = (const float*)d_in[9];
  const float* Wim  = (const float*)d_in[10];

  char* w = (char*)d_ws;
  const size_t BN = (size_t)B * N;
  float* sir = (float*)w; w += BN * 4;
  float* sii = (float*)w; w += BN * 4;
  float* sjr = (float*)w; w += BN * 4;
  float* sji = (float*)w; w += BN * 4;
  float* mxr = (float*)w; w += BN * 4;
  float* ivr = (float*)w; w += BN * 4;
  float* mxi = (float*)w; w += BN * 4;
  float* ivi = (float*)w; w += BN * 4;
  unsigned short* WrT = (unsigned short*)w; w += (size_t)K1 * CO * C * 2;
  unsigned short* WiT = (unsigned short*)w; w += (size_t)K1 * CO * C * 2;
  unsigned short* P1T = (unsigned short*)w; w += (size_t)K1 * B * CO * N * 2;
  unsigned short* P2T = (unsigned short*)w; w += (size_t)K1 * B * CO * N * 2;

  float* outr = (float*)d_out;
  float* outi = outr + (size_t)B * N * CO;

  kern_wt<<<(K1 * CO * C + 255) / 256, 256, 0, stream>>>(Wre, Wim, WrT, WiT);
  kern_s<<<(B * N + 255) / 256, 256, 0, stream>>>(Xr, Xi, War, Wai, sir, sii, sjr, sji);
  kern_stats<<<(B * N) / 8, 256, 0, stream>>>(sir, sii, sjr, sji, bar, bai, mb,
                                              mxr, ivr, mxi, ivi);
  kern_p<<<(K1 * B * (N / 16)) / 8, 256, 0, stream>>>(Xr, Xi, WrT, WiT, P1T, P2T);
  kern_main<<<dim3(B, N / 64), 128, 0, stream>>>(lapr, lapi, sir, sii, sjr, sji,
                                                 mxr, ivr, mxi, ivi, bar, bai, mb,
                                                 P1T, P2T, outr, outi);
}
